// ParallelEmbedding_54666343743646
// MI455X (gfx1250) — compile-verified, matrix operand order checked by reading, not run on hardware
//
#include <hip/hip_runtime.h>
#include <stdint.h>

// Problem constants (from the reference)
#define VOCAB 128000
#define DIM   1024
#define RANK  256

typedef __attribute__((ext_vector_type(16))) __bf16 v16bf;
typedef __attribute__((ext_vector_type(8)))  float  v8f;

__device__ __forceinline__ uint16_t f32_to_bf16(float f) {
  uint32_t u = __builtin_bit_cast(uint32_t, f);
  uint32_t lsb = (u >> 16) & 1u;
  u += 0x7FFFu + lsb;              // round-to-nearest-even
  return (uint16_t)(u >> 16);
}

// K position within a 32-wide K-chunk for fragment slot i (0..15) and lane
// group (0 = lanes 0-15, 1 = lanes 16-31), per the CDNA5 16-bit A/B fragment
// layout (ISA 7.12.2).
__device__ __forceinline__ int klocal(int i, int group) {
  int v = i >> 1, h = i & 1;
  int k = (v < 4) ? (2 * v + h) : (16 + 2 * (v - 4) + h);
  return k + 8 * group;
}

// Repack B [RANK x DIM] fp32 -> bf16 fragments in ws:
// layout [ntile(64)][kchunk(8)][lane(32)][i(16)] so each (ntile,kchunk,lane)
// fragment is 32 contiguous bytes (two b128 loads).
__global__ void convB_kernel(const float* __restrict__ B,
                             uint16_t* __restrict__ Bb) {
  int f = blockIdx.x * blockDim.x + threadIdx.x;   // 0 .. RANK*DIM-1
  int i    = f & 15;
  int lane = (f >> 4) & 31;
  int kc   = (f >> 9) & 7;
  int nt   = f >> 12;
  int group = lane >> 4;
  int n  = nt * 16 + (lane & 15);
  int kk = kc * 32 + klocal(i, group);
  Bb[f] = f32_to_bf16(B[kk * DIM + n]);
}

__global__ __launch_bounds__(256)
void lora_embed_kernel(const int* __restrict__ x,
                       const float* __restrict__ W,
                       const float* __restrict__ A,
                       const uint16_t* __restrict__ Bb,
                       float* __restrict__ out) {
  __shared__ int tok[16];
  __shared__ __align__(32) uint16_t sA[8 * 32 * 16];  // 8 KB: A fragments

  const int tile = blockIdx.x;        // one 16-token M-tile per block
  const int t0   = tile * 16;
  const int tid  = threadIdx.x;

  if (tid < 16) tok[tid] = x[t0 + tid];
  __syncthreads();

  // Gather 16 A rows (coalesced 64B per thread, streaming/NT) and scatter
  // into the WMMA A-fragment layout in LDS as bf16.
  {
    const int m     = tid >> 4;             // token row 0..15
    const int kbase = (tid & 15) * 16;      // 16 consecutive K per thread
    const float* Arow = A + (size_t)tok[m] * RANK + kbase;
#pragma unroll
    for (int j = 0; j < 16; ++j) {
      float av = __builtin_nontemporal_load(Arow + j);
      int K  = kbase + j;
      int kc = K >> 5;
      int kl = K & 31;
      int g  = (kl >> 3) & 1;
      int q  = kl - 8 * g;
      int v, h;
      if (q < 8) { v = q >> 1;              h = q & 1; }
      else       { v = 4 + ((q - 16) >> 1); h = q & 1; }
      int i    = 2 * v + h;
      int lane = m + 16 * g;
      sA[(kc * 32 + lane) * 16 + i] = f32_to_bf16(av);
    }
  }
  __syncthreads();

  const int wave = tid >> 5;   // 0..7
  const int lane = tid & 31;
  const int lrow = lane & 15;
  const int lgrp = lane >> 4;

  // ---- C preload: all 8 N-tiles' weight values in one up-front NT batch ----
  // C/D layout: VGPR r, lanes 0-15 -> M=r, lanes 16-31 -> M=r+8.
  // acc[i] is the accumulator for N-tile nt = wave + 8*i.
  v8f acc[8];
#pragma unroll
  for (int r = 0; r < 8; ++r) {
    const float* wb = W + (size_t)tok[r + 8 * lgrp] * DIM + wave * 16 + lrow;
#pragma unroll
    for (int i = 0; i < 8; ++i)
      acc[i][r] = __builtin_nontemporal_load(wb + i * 128);
  }

  float* obase = out + (size_t)(t0 + 8 * lgrp) * DIM + wave * 16 + lrow;

  // Fragment (nt = wave + 8i, k) lives at v16bf index (i*64 + k)*32 from bbase.
  const v16bf* bbase =
      reinterpret_cast<const v16bf*>(Bb) + (wave * 8) * 32 + lane;

  // ---- 64 WMMA steps, k-outer / nt-inner ----
  // B: 4-deep software pipeline in distinct registers.  A: ping-pong double
  // buffer from LDS, prefetched 7 steps ahead (separate DScnt counter).
  // __builtin_amdgcn_sched_barrier(0) pins the written order so the pre-RA
  // scheduler cannot sink the prefetch loads back to distance-1 (which is
  // what collapsed the pipeline in the previous build).
  v16bf buf[4];
#pragma unroll
  for (int p = 0; p < 4; ++p)       // prologue: steps 0..3 are k=0, i=p
    buf[p] = bbase[(p * 64 + 0) * 32];

  v16bf afr[2];
  afr[0] = *reinterpret_cast<const v16bf*>(&sA[(0 * 32 + lane) * 16]);
  __builtin_amdgcn_sched_barrier(0);

#pragma unroll
  for (int s = 0; s < 64; ++s) {
    const int k = s >> 3;
    const int i = s & 7;

    acc[i] = __builtin_amdgcn_wmma_f32_16x16x32_bf16(
        /*neg_a=*/false, afr[k & 1], /*neg_b=*/false, buf[s & 3],
        /*c_mod=*/(short)0, acc[i], /*reuse_a=*/false, /*reuse_b=*/false);

    // Prefetch B fragment for step s+4 into the slot just freed.
    if (s + 4 < 64) {
      const int s2 = s + 4;
      buf[s & 3] = bbase[(((s2 & 7) * 64) + (s2 >> 3)) * 32];
    }
    // Prefetch next K-chunk's A fragment early in each chunk (distance 7).
    if (i == 1 && k < 7)
      afr[(k + 1) & 1] =
          *reinterpret_cast<const v16bf*>(&sA[((k + 1) * 32 + lane) * 16]);

    __builtin_amdgcn_sched_barrier(0);
  }

  // ---- Streaming fp32 stores ----
#pragma unroll
  for (int i = 0; i < 8; ++i) {
#pragma unroll
    for (int r = 0; r < 8; ++r)
      __builtin_nontemporal_store(acc[i][r], obase + i * 128 + r * DIM);
  }
}

extern "C" void kernel_launch(void* const* d_in, const int* in_sizes, int n_in,
                              void* d_out, int out_size, void* d_ws, size_t ws_size,
                              hipStream_t stream) {
  const int*   x = (const int*)d_in[0];
  const float* W = (const float*)d_in[1];
  const float* A = (const float*)d_in[2];
  const float* B = (const float*)d_in[3];
  float* out = (float*)d_out;
  uint16_t* Bb = (uint16_t*)d_ws;   // 512 KB bf16 repacked B

  // 1) Repack B into bf16 WMMA fragments (L2-resident afterwards).
  const int belems = RANK * DIM;                 // 262144
  convB_kernel<<<belems / 256, 256, 0, stream>>>(B, Bb);

  // 2) Fused gather + LoRA matmul + add.
  const int ntok = in_sizes[0];                  // 16384
  lora_embed_kernel<<<ntok / 16, 256, 0, stream>>>(x, W, A, Bb, out);
}
